// LWGA_4707284156549
// MI455X (gfx1250) — matrix operand order repaired and checked by reading, not compiled, not run on hardware
//
#include <hip/hip_runtime.h>
#include <math.h>

#define B_N 16
#define CDIM 256
#define HS 96
#define PLANE 9216          // 96*96
#define DSC 64
#define NPIX (B_N*PLANE)    // 147456
#define PLANE2 2304         // 48*48
#define NPIX2 (B_N*PLANE2)  // 36864
#define PLANE3 1024         // 32*32

typedef unsigned short u16;
typedef __attribute__((ext_vector_type(16))) __bf16 v16bf;
typedef __attribute__((ext_vector_type(8)))  float  v8f;

union AF { v16bf v; u16 s[16]; };

__device__ __forceinline__ u16 f2bf(float f) {
  union { float f; unsigned u; } x; x.f = f;
  unsigned r = x.u + 0x7FFFu + ((x.u >> 16) & 1u);   // RNE
  return (u16)(r >> 16);
}
__device__ __forceinline__ float geluf(float v) {
  return 0.5f * v * (1.0f + erff(v * 0.7071067811865476f));
}
__device__ __forceinline__ float sigm(float v) { return 1.0f / (1.0f + expf(-v)); }

// One 16x16 output tile: fully-unrolled K-loop of v_wmma_f32_16x16x32_bf16.
// WPK: A-fragments prepacked, lane half h holds k 16h..16h+15 ([tile][lane][16 halves], 32B each).
// LDSARR: bf16 activations, [pixLocal][CC] row-major; B-frag: lanes 0-15 read
// K 0..15 (col = PT*16+lane), lanes 16-31 read K 16..31 (same contiguous k order as the A prepack).
// KT is a compile-time constant at every use site so this unrolls and the
// prefetch guard constant-folds.
#define WMMA_TILE(ACC, WPK, MT, PT, KT, LDSARR, CC)                                     \
  do {                                                                                  \
    int lane_ = (int)(threadIdx.x & 31u);                                               \
    int col_  = (PT) * 16 + (lane_ & 15);                                               \
    int hi_   = lane_ >> 4;                                                             \
    _Pragma("unroll")                                                                   \
    for (int kt_ = 0; kt_ < (KT); ++kt_) {                                              \
      AF a_, b_;                                                                        \
      a_.v = *(const v16bf*)((WPK) + (size_t)((((MT)*(KT) + kt_) * 32 + lane_) << 4));  \
      if (kt_ + 1 < (KT))                                                               \
        __builtin_prefetch((const void*)((WPK) +                                        \
            (size_t)((((MT)*(KT) + kt_ + 1) * 32 + lane_) << 4)), 0, 3);                \
      b_.v = *(const v16bf*)(&(LDSARR)[(size_t)col_ * (CC) + kt_ * 32 + (hi_ << 4)]);   \
      (ACC) = __builtin_amdgcn_wmma_f32_16x16x32_bf16(                                  \
          false, a_.v, false, b_.v, (short)0, (ACC), false, false);                     \
    }                                                                                   \
  } while (0)

// ---------------- weight prepack: f32 (Cout,Cin[,stride]) -> A-frag bf16 ---------
__global__ void k_prepack(const float* __restrict__ w, u16* __restrict__ dst,
                          int cout, int cin, int wstride, int woff) {
  int d = blockIdx.x * 256 + threadIdx.x;
  int total = cout * cin;
  if (d >= total) return;
  int h = d & 15;
  int lane = (d >> 4) & 31;
  int tile = d >> 9;
  int KT = cin >> 5;
  int kt = tile % KT;
  int mt = tile / KT;
  int m = mt * 16 + (lane & 15);
  int k = kt * 32 + h + ((lane >> 4) << 4);
  dst[d] = f2bf(w[(size_t)(m * cin + k) * wstride + woff]);
}

// ---------------- generic WMMA 1x1-conv (optional 2-src concat, bias, gelu) ------
// CIN templated -> compile-time KT, exact-size LDS tile, fully unrolled WMMA body.
template <int CIN>
__global__ void k_gemm1x1(const float* __restrict__ s0, long long sb0, int csplit,
                          const float* __restrict__ s1, long long sb1,
                          const u16* __restrict__ wpk,
                          const float* __restrict__ bias, int act,
                          float* __restrict__ out, long long outb,
                          int cout, int plane) {
  __shared__ u16 lds[CIN * 64];
  int tid = threadIdx.x;
  long long colBase = (long long)blockIdx.x * 64;
  for (int e = tid; e < (CIN << 6); e += 256) {
    int c = e >> 6, p = e & 63;
    long long col = colBase + p;
    int b = (int)(col / plane);
    int pp = (int)(col - (long long)b * plane);
    float v = (c < csplit) ? s0[(size_t)b * sb0 + (size_t)c * plane + pp]
                           : s1[(size_t)b * sb1 + (size_t)(c - csplit) * plane + pp];
    lds[p * CIN + c] = f2bf(v);
  }
  __syncthreads();
  int wave = tid >> 5, lane = tid & 31;
  int n = lane & 15, hi = lane >> 4;
  int jobs = (cout >> 4) << 2;
  for (int job = wave; job < jobs; job += 8) {
    int mt = job >> 2, pt = job & 3;
    v8f acc = {0, 0, 0, 0, 0, 0, 0, 0};
    WMMA_TILE(acc, wpk, mt, pt, (CIN / 32), lds, CIN);
    long long gcol = colBase + pt * 16 + n;
    int b = (int)(gcol / plane);
    int pp = (int)(gcol - (long long)b * plane);
#pragma unroll
    for (int r = 0; r < 8; ++r) {
      int row = mt * 16 + r + hi * 8;
      float v = acc[r];
      if (bias) v += bias[row];
      if (act) v = geluf(v);
      out[(size_t)b * outb + (size_t)row * plane + pp] = v;
    }
  }
}

// ---------------- fused pixel-attention branch: 64->256 (bn+gelu) ->64, gate -----
__global__ void k_pa(const float* __restrict__ X,
                     const u16* __restrict__ pk1, const u16* __restrict__ pk2,
                     const float* __restrict__ bng, const float* __restrict__ bnb,
                     float* __restrict__ x1pa, float* __restrict__ xa) {
  __shared__ u16 ldsx[64 * 64];
  __shared__ u16 ldsh[64 * 256];
  int tid = threadIdx.x;
  long long colBase = (long long)blockIdx.x * 64;
  for (int e = tid; e < 4096; e += 256) {
    int c = e >> 6, p = e & 63;
    long long col = colBase + p;
    int b = (int)(col / PLANE);
    int pp = (int)(col - (long long)b * PLANE);
    ldsx[p * 64 + c] = f2bf(X[((size_t)b * CDIM + c) * PLANE + pp]);
  }
  __syncthreads();
  int wave = tid >> 5, lane = tid & 31, n = lane & 15, hi = lane >> 4;
  float rs = rsqrtf(1.00001f);
  for (int j = 0; j < 8; ++j) {              // stage1: 16 cout tiles x 4 pix tiles
    int job = wave + j * 8;
    int mt = job >> 2, pt = job & 3;
    v8f acc = {0, 0, 0, 0, 0, 0, 0, 0};
    WMMA_TILE(acc, pk1, mt, pt, 2, ldsx, 64);
    int colL = pt * 16 + n;
#pragma unroll
    for (int r = 0; r < 8; ++r) {
      int row = mt * 16 + r + hi * 8;
      float v = geluf(acc[r] * bng[row] * rs + bnb[row]);
      ldsh[colL * 256 + row] = f2bf(v);
    }
  }
  __syncthreads();
  for (int j = 0; j < 2; ++j) {              // stage2: 4 cout tiles x 4 pix tiles
    int job = wave + j * 8;
    int mt = job >> 2, pt = job & 3;
    v8f acc = {0, 0, 0, 0, 0, 0, 0, 0};
    WMMA_TILE(acc, pk2, mt, pt, 8, ldsh, 256);
    long long gcol = colBase + pt * 16 + n;
    int b = (int)(gcol / PLANE);
    int pp = (int)(gcol - (long long)b * PLANE);
#pragma unroll
    for (int r = 0; r < 8; ++r) {
      int row = mt * 16 + r + hi * 8;
      float x1v = X[((size_t)b * CDIM + row) * PLANE + pp];
      float pa = x1v * sigm(acc[r]);
      x1pa[((size_t)b * DSC + row) * PLANE + pp] = pa;
      xa[((size_t)b * CDIM + row) * PLANE + pp] = x1v + pa;
    }
  }
}

// ---------------- 3x3 conv 64->64 as 9 accumulated shifted WMMA GEMMs ------------
__global__ void k_la(const float* __restrict__ x2f, const u16* __restrict__ lapk,
                     const float* __restrict__ lag, const float* __restrict__ lab,
                     float* __restrict__ xa) {
  __shared__ u16 lds[64 * 64];
  int tid = threadIdx.x;
  long long colBase = (long long)blockIdx.x * 64;
  int wave = tid >> 5, lane = tid & 31, n = lane & 15, hi = lane >> 4;
  int mt0 = wave >> 2, pt = wave & 3, mt1 = mt0 + 2;
  v8f acc0 = {0, 0, 0, 0, 0, 0, 0, 0};
  v8f acc1 = {0, 0, 0, 0, 0, 0, 0, 0};
  for (int tap = 0; tap < 9; ++tap) {
    int dy = tap / 3 - 1, dx = tap % 3 - 1;
    for (int e = tid; e < 4096; e += 256) {
      int c = e >> 6, p = e & 63;
      long long col = colBase + p;
      int b = (int)(col / PLANE);
      int pp = (int)(col - (long long)b * PLANE);
      int row = pp / HS, cc = pp % HS;
      int ii = row + dy, jj = cc + dx;
      float v = 0.f;
      if ((unsigned)ii < (unsigned)HS && (unsigned)jj < (unsigned)HS)
        v = x2f[((size_t)b * DSC + c) * PLANE + ii * HS + jj];
      lds[p * 64 + c] = f2bf(v);
    }
    __syncthreads();
    const u16* wp = lapk + (size_t)tap * 4096;
    WMMA_TILE(acc0, wp, mt0, pt, 2, lds, 64);
    WMMA_TILE(acc1, wp, mt1, pt, 2, lds, 64);
    __syncthreads();
  }
  float rs = rsqrtf(1.00001f);
  long long gcol = colBase + pt * 16 + n;
  int b = (int)(gcol / PLANE);
  int pp = (int)(gcol - (long long)b * PLANE);
#pragma unroll
  for (int r = 0; r < 8; ++r) {
    int row = mt0 * 16 + r + hi * 8;
    float v = geluf(acc0[r] * lag[row] * rs + lab[row]);
    xa[((size_t)b * CDIM + 64 + row) * PLANE + pp] = v;
  }
#pragma unroll
  for (int r = 0; r < 8; ++r) {
    int row = mt1 * 16 + r + hi * 8;
    float v = geluf(acc1[r] * lag[row] * rs + lab[row]);
    xa[((size_t)b * CDIM + 64 + row) * PLANE + pp] = v;
  }
}

// ---------------- fused MLP: 256->1024 (bn+gelu in 128KB LDS) ->256 + residual ---
__global__ void k_mlp(const float* __restrict__ xa, const float* __restrict__ X,
                      const u16* __restrict__ pk1, const u16* __restrict__ pk2,
                      const float* __restrict__ bng, const float* __restrict__ bnb,
                      const float* __restrict__ n1g, const float* __restrict__ n1b,
                      float* __restrict__ out) {
  extern __shared__ u16 sm[];
  u16* ldsa = sm;               // 64 px * 256 ch  (32 KB)
  u16* ldsh = sm + 64 * 256;    // 64 px * 1024 ch (128 KB)
  int tid = threadIdx.x;
  long long colBase = (long long)blockIdx.x * 64;
  for (int e = tid; e < 16384; e += 256) {
    int c = e >> 6, p = e & 63;
    long long col = colBase + p;
    int b = (int)(col / PLANE);
    int pp = (int)(col - (long long)b * PLANE);
    ldsa[p * 256 + c] = f2bf(xa[((size_t)b * CDIM + c) * PLANE + pp]);
  }
  __syncthreads();
  int wave = tid >> 5, lane = tid & 31, n = lane & 15, hi = lane >> 4;
  float rs = rsqrtf(1.00001f);
  for (int j = 0; j < 32; ++j) {             // stage1: 64 cout tiles x 4 pix tiles
    int job = wave + j * 8;
    int mt = job >> 2, pt = job & 3;
    v8f acc = {0, 0, 0, 0, 0, 0, 0, 0};
    WMMA_TILE(acc, pk1, mt, pt, 8, ldsa, 256);
    int colL = pt * 16 + n;
#pragma unroll
    for (int r = 0; r < 8; ++r) {
      int row = mt * 16 + r + hi * 8;
      float v = geluf(acc[r] * bng[row] * rs + bnb[row]);
      ldsh[colL * 1024 + row] = f2bf(v);
    }
  }
  __syncthreads();
  for (int j = 0; j < 8; ++j) {              // stage2: 16 cout tiles x 4 pix tiles
    int job = wave + j * 8;
    int mt = job >> 2, pt = job & 3;
    v8f acc = {0, 0, 0, 0, 0, 0, 0, 0};
    WMMA_TILE(acc, pk2, mt, pt, 32, ldsh, 1024);
    long long gcol = colBase + pt * 16 + n;
    int b = (int)(gcol / PLANE);
    int pp = (int)(gcol - (long long)b * PLANE);
#pragma unroll
    for (int r = 0; r < 8; ++r) {
      int row = mt * 16 + r + hi * 8;
      size_t gi = ((size_t)b * CDIM + row) * PLANE + pp;
      out[gi] = X[gi] + (acc[r] * n1g[row] * rs + n1b[row]);
    }
  }
}

// ================= small VALU kernels ============================================
__global__ void k_mp3(const float* __restrict__ X, float* __restrict__ o) {
  int i = blockIdx.x * 256 + threadIdx.x;
  if (i >= B_N * DSC * PLANE) return;
  int pp = i % PLANE, bc = i / PLANE;
  int c = bc & 63, b = bc >> 6;
  int y = pp / HS, x = pp % HS;
  const float* s = X + ((size_t)b * CDIM + 128 + c) * PLANE;
  float m = -1e30f;
  for (int dy = -1; dy <= 1; ++dy) {
    int yy = y + dy; if ((unsigned)yy >= (unsigned)HS) continue;
    for (int dx = -1; dx <= 1; ++dx) {
      int xx = x + dx; if ((unsigned)xx >= (unsigned)HS) continue;
      float v = s[yy * HS + xx]; m = v > m ? v : m;
    }
  }
  o[i] = m;
}

__global__ void k_blur(const float* __restrict__ m, float* __restrict__ t) {
  int i = blockIdx.x * 256 + threadIdx.x;
  if (i >= B_N * DSC * PLANE3) return;
  int x = i & 31, y = (i >> 5) & 31, bc = i >> 10;
  const float* s = m + (size_t)bc * PLANE;
  const float wv[4] = {1.f, 3.f, 3.f, 1.f};
  float acc = 0.f;
  for (int r = 0; r < 4; ++r) {
    int py = 3 * y + r - 1;
    int sy = py < 0 ? -py : (py > 95 ? 190 - py : py);
    for (int q = 0; q < 4; ++q) {
      int px = 3 * x + q - 1;
      int sx = px < 0 ? -px : (px > 95 ? 190 - px : px);
      acc += wv[r] * wv[q] * s[sy * HS + sx];
    }
  }
  t[i] = acc * (1.0f / 64.0f);
}

__global__ void k_hv1(const float* __restrict__ t, const float* __restrict__ h1,
                      const float* __restrict__ v1, float* __restrict__ g) {
  int i = blockIdx.x * 256 + threadIdx.x;
  if (i >= B_N * DSC * PLANE3) return;
  int x = i & 31, y = (i >> 5) & 31, bc = i >> 10, c = bc & 63;
  const float* s = t + (size_t)bc * PLANE3;
  float acc = 0.f;
  for (int ty = 0; ty < 7; ++ty) {
    int yy = y + ty - 3; if ((unsigned)yy >= 32u) continue;
    for (int tx = 0; tx < 3; ++tx) {
      int xx = x + tx - 1; if ((unsigned)xx >= 32u) continue;
      acc += h1[(c * 7 + ty) * 3 + tx] * s[yy * 32 + xx];
    }
  }
  for (int ty = 0; ty < 3; ++ty) {
    int yy = y + ty - 1; if ((unsigned)yy >= 32u) continue;
    for (int tx = 0; tx < 7; ++tx) {
      int xx = x + tx - 3; if ((unsigned)xx >= 32u) continue;
      acc += v1[(c * 3 + ty) * 7 + tx] * s[yy * 32 + xx];
    }
  }
  g[i] = acc;
}

__global__ void k_htr(const float* __restrict__ t, float* __restrict__ ht) {
  int i = blockIdx.x * 256 + threadIdx.x;
  if (i >= B_N * DSC * 2016) return;
  int j = i % 63, rest = i / 63;
  int y = rest & 31, bc = rest >> 5;
  int f = y * 63 + j, r = f >> 6, cc = f & 63;
  ht[i] = (cc < 32) ? t[(size_t)bc * PLANE3 + r * 32 + cc] : 0.f;
}

__global__ void k_hconv(const float* __restrict__ ht, const float* __restrict__ w,
                        float* __restrict__ o) {
  int i = blockIdx.x * 256 + threadIdx.x;
  if (i >= B_N * DSC * 2016) return;
  int x = i % 63, rest = i / 63;
  int y = rest & 31, bc = rest >> 5, c = bc & 63;
  const float* s = ht + (size_t)bc * 2016;
  float acc = 0.f;
  for (int ty = 0; ty < 7; ++ty) {
    int yy = y + ty - 3; if ((unsigned)yy >= 32u) continue;
    for (int tx = 0; tx < 3; ++tx) {
      int xx = x + tx - 1; if ((unsigned)xx >= 63u) continue;
      acc += w[(c * 7 + ty) * 3 + tx] * s[yy * 63 + xx];
    }
  }
  o[i] = acc;
}

__global__ void k_hinv(const float* __restrict__ hc, float* __restrict__ g) {
  int i = blockIdx.x * 256 + threadIdx.x;
  if (i >= B_N * DSC * PLANE3) return;
  int j = i & 31, y = (i >> 5) & 31, bc = i >> 10;
  int f = y * 64 + j;   // (32,63) row-major flat index == f
  g[i] += hc[(size_t)bc * 2016 + f];
}

__global__ void k_vtr(const float* __restrict__ t, float* __restrict__ vt) {
  int i = blockIdx.x * 256 + threadIdx.x;
  if (i >= B_N * DSC * 2016) return;
  int q = i & 31, rest = i >> 5;
  int p = rest % 63, bc = rest / 63;
  int f = q * 63 + p, r = f >> 6, cc = f & 63;
  vt[i] = (cc < 32) ? t[(size_t)bc * PLANE3 + cc * 32 + r] : 0.f;
}

__global__ void k_vconv(const float* __restrict__ vt, const float* __restrict__ w,
                        float* __restrict__ o) {
  int i = blockIdx.x * 256 + threadIdx.x;
  if (i >= B_N * DSC * 2016) return;
  int x = i & 31, rest = i >> 5;
  int y = rest % 63, bc = rest / 63, c = bc & 63;
  const float* s = vt + (size_t)bc * 2016;
  float acc = 0.f;
  for (int ty = 0; ty < 3; ++ty) {
    int yy = y + ty - 1; if ((unsigned)yy >= 63u) continue;
    for (int tx = 0; tx < 7; ++tx) {
      int xx = x + tx - 3; if ((unsigned)xx >= 32u) continue;
      acc += w[(c * 3 + ty) * 7 + tx] * s[yy * 32 + xx];
    }
  }
  o[i] = acc;
}

__global__ void k_vinv(const float* __restrict__ vc, float* __restrict__ g) {
  int i = blockIdx.x * 256 + threadIdx.x;
  if (i >= B_N * DSC * PLANE3) return;
  int j = i & 31, y = (i >> 5) & 31, bc = i >> 10;
  int f = j * 64 + y;
  g[i] += vc[(size_t)bc * 2016 + (f % 63) * 32 + (f / 63)];
}

__global__ void k_gate(float* __restrict__ g, const float* __restrict__ gg,
                       const float* __restrict__ gb) {
  int i = blockIdx.x * 256 + threadIdx.x;
  if (i >= B_N * DSC * PLANE3) return;
  int c = (i >> 10) & 63;
  float rs = rsqrtf(1.00001f);
  g[i] = sigm(g[i] * gg[c] * rs + gb[c]);
}

__global__ void k_mra(const float* __restrict__ X, const float* __restrict__ gate,
                      float* __restrict__ x3m, float* __restrict__ xa) {
  int i = blockIdx.x * 256 + threadIdx.x;
  if (i >= B_N * DSC * PLANE) return;
  int pp = i % PLANE, bc = i / PLANE;
  int c = bc & 63, b = bc >> 6;
  int y = pp / HS, x = pp % HS;
  float gv = gate[((size_t)bc * 32 + y / 3) * 32 + x / 3];
  float xv = X[((size_t)b * CDIM + 128 + c) * PLANE + pp];
  float m = xv * gv;
  x3m[i] = m;
  xa[((size_t)b * CDIM + 128 + c) * PLANE + pp] = xv + m;
}

__global__ void k_mp2(const float* __restrict__ x4f, float* __restrict__ xp,
                      int* __restrict__ idx) {
  int i = blockIdx.x * 256 + threadIdx.x;
  if (i >= B_N * DSC * PLANE2) return;
  int p = i % PLANE2, bc = i / PLANE2;
  int j2 = p % 48, i2 = p / 48;
  const float* s = x4f + (size_t)bc * PLANE;
  float v0 = s[(2 * i2) * HS + 2 * j2];
  float v1 = s[(2 * i2) * HS + 2 * j2 + 1];
  float v2 = s[(2 * i2 + 1) * HS + 2 * j2];
  float v3 = s[(2 * i2 + 1) * HS + 2 * j2 + 1];
  float best = v0; int k = 0;
  if (v1 > best) { best = v1; k = 1; }
  if (v2 > best) { best = v2; k = 2; }
  if (v3 > best) { best = v3; k = 3; }
  xp[i] = best; idx[i] = k;
}

__global__ void k_dw5(const float* __restrict__ in, const float* __restrict__ w,
                      const float* __restrict__ bias, float* __restrict__ o) {
  int i = blockIdx.x * 256 + threadIdx.x;
  if (i >= B_N * DSC * PLANE2) return;
  int p = i % PLANE2, bc = i / PLANE2, c = bc & 63;
  int x = p % 48, y = p / 48;
  const float* s = in + (size_t)bc * PLANE2;
  float acc = bias[c];
  for (int ty = 0; ty < 5; ++ty) {
    int yy = y + ty - 2; if ((unsigned)yy >= 48u) continue;
    for (int tx = 0; tx < 5; ++tx) {
      int xx = x + tx - 2; if ((unsigned)xx >= 48u) continue;
      acc += w[(c * 5 + ty) * 5 + tx] * s[yy * 48 + xx];
    }
  }
  o[i] = acc;
}

__global__ void k_dw7(const float* __restrict__ in, const float* __restrict__ w,
                      const float* __restrict__ bias, float* __restrict__ o) {
  int i = blockIdx.x * 256 + threadIdx.x;
  if (i >= B_N * DSC * PLANE2) return;
  int p = i % PLANE2, bc = i / PLANE2, c = bc & 63;
  int x = p % 48, y = p / 48;
  const float* s = in + (size_t)bc * PLANE2;
  float acc = bias[c];
  for (int ty = 0; ty < 7; ++ty) {
    int yy = y + (ty - 3) * 3; if ((unsigned)yy >= 48u) continue;
    for (int tx = 0; tx < 7; ++tx) {
      int xx = x + (tx - 3) * 3; if ((unsigned)xx >= 48u) continue;
      acc += w[(c * 7 + ty) * 7 + tx] * s[yy * 48 + xx];
    }
  }
  o[i] = acc;
}

__global__ void k_agg(const float* __restrict__ a1c, const float* __restrict__ a2c,
                      float* __restrict__ agg) {
  int i = blockIdx.x * 256 + threadIdx.x;
  if (i >= B_N * PLANE2) return;
  int p = i % PLANE2, b = i / PLANE2;
  float sum = 0.f, mx = -1e30f;
  for (int c = 0; c < 32; ++c) {
    float v = a1c[((size_t)b * 32 + c) * PLANE2 + p];
    sum += v; mx = v > mx ? v : mx;
  }
  for (int c = 0; c < 32; ++c) {
    float v = a2c[((size_t)b * 32 + c) * PLANE2 + p];
    sum += v; mx = v > mx ? v : mx;
  }
  agg[((size_t)b * 2 + 0) * PLANE2 + p] = sum * (1.0f / 64.0f);
  agg[((size_t)b * 2 + 1) * PLANE2 + p] = mx;
}

__global__ void k_sq(const float* __restrict__ agg, const float* __restrict__ w,
                     const float* __restrict__ bias, float* __restrict__ sig) {
  int i = blockIdx.x * 256 + threadIdx.x;
  if (i >= B_N * 2 * PLANE2) return;
  int p = i % PLANE2, rest = i / PLANE2;
  int o = rest & 1, b = rest >> 1;
  int x = p % 48, y = p / 48;
  float acc = bias[o];
  for (int ic = 0; ic < 2; ++ic) {
    const float* s = agg + ((size_t)b * 2 + ic) * PLANE2;
    for (int ty = 0; ty < 7; ++ty) {
      int yy = y + ty - 3; if ((unsigned)yy >= 48u) continue;
      for (int tx = 0; tx < 7; ++tx) {
        int xx = x + tx - 3; if ((unsigned)xx >= 48u) continue;
        acc += w[((o * 2 + ic) * 7 + ty) * 7 + tx] * s[yy * 48 + xx];
      }
    }
  }
  sig[i] = sigm(acc);
}

__global__ void k_amix(const float* __restrict__ a1c, const float* __restrict__ a2c,
                       const float* __restrict__ sig, float* __restrict__ amix) {
  int i = blockIdx.x * 256 + threadIdx.x;
  if (i >= B_N * 32 * PLANE2) return;
  int p = i % PLANE2, rest = i / PLANE2;
  int b = rest >> 5;
  float s0 = sig[((size_t)b * 2 + 0) * PLANE2 + p];
  float s1 = sig[((size_t)b * 2 + 1) * PLANE2 + p];
  amix[i] = a1c[i] * s0 + a2c[i] * s1;
}

__global__ void k_prod(const float* __restrict__ a, const float* __restrict__ b,
                       float* __restrict__ o) {
  int i = blockIdx.x * 256 + threadIdx.x;
  if (i >= B_N * DSC * PLANE2) return;
  o[i] = a[i] * b[i];
}

__global__ void k_unpool(const float* __restrict__ yb, const int* __restrict__ idx,
                         const float* __restrict__ X, const float* __restrict__ g4,
                         const float* __restrict__ b4, float* __restrict__ xa) {
  int i = blockIdx.x * 256 + threadIdx.x;
  if (i >= B_N * DSC * PLANE2) return;
  int p = i % PLANE2, bc = i / PLANE2;
  int c = bc & 63, b = bc >> 6;
  int j2 = p % 48, i2 = p / 48;
  float yv = yb[i];
  int k = idx[i];
  float rs = rsqrtf(1.00001f);
  float sc = g4[c] * rs, bo = b4[c];
  const float* xs = X + ((size_t)b * CDIM + 192 + c) * PLANE;
  float* os = xa + ((size_t)b * CDIM + 192 + c) * PLANE;
  for (int dy = 0; dy < 2; ++dy)
    for (int dx = 0; dx < 2; ++dx) {
      int op = (2 * i2 + dy) * HS + (2 * j2 + dx);
      float v = (dy * 2 + dx == k) ? yv : 0.f;
      os[op] = (xs[op] + v) * sc + bo;
    }
}

// ================= host orchestration ===========================================
extern "C" void kernel_launch(void* const* d_in, const int* in_sizes, int n_in,
                              void* d_out, int out_size, void* d_ws, size_t ws_size,
                              hipStream_t stream) {
  (void)in_sizes; (void)n_in; (void)out_size; (void)ws_size;
  const float* X       = (const float*)d_in[0];
  const float* pa_w1   = (const float*)d_in[1];
  const float* pa_bn_g = (const float*)d_in[2];
  const float* pa_bn_b = (const float*)d_in[3];
  const float* pa_w2   = (const float*)d_in[4];
  const float* fuse1_w = (const float*)d_in[5];
  const float* la_w    = (const float*)d_in[6];
  const float* la_bn_g = (const float*)d_in[7];
  const float* la_bn_b = (const float*)d_in[8];
  const float* h1_w    = (const float*)d_in[9];
  const float* v1_w    = (const float*)d_in[10];
  const float* h2_w    = (const float*)d_in[11];
  const float* v2_w    = (const float*)d_in[12];
  const float* mra_g   = (const float*)d_in[13];
  const float* mra_b   = (const float*)d_in[14];
  const float* fuse2_w = (const float*)d_in[15];
  const float* g_p1_w  = (const float*)d_in[16];
  const float* g_p1_b  = (const float*)d_in[17];
  const float* g_c0_w  = (const float*)d_in[18];
  const float* g_c0_b  = (const float*)d_in[19];
  const float* g_cs_w  = (const float*)d_in[20];
  const float* g_cs_b  = (const float*)d_in[21];
  const float* g_c1_w  = (const float*)d_in[22];
  const float* g_c1_b  = (const float*)d_in[23];
  const float* g_c2_w  = (const float*)d_in[24];
  const float* g_c2_b  = (const float*)d_in[25];
  const float* g_sq_w  = (const float*)d_in[26];
  const float* g_sq_b  = (const float*)d_in[27];
  const float* g_cv_w  = (const float*)d_in[28];
  const float* g_cv_b  = (const float*)d_in[29];
  const float* g_p2_w  = (const float*)d_in[30];
  const float* g_p2_b  = (const float*)d_in[31];
  const float* n4_g    = (const float*)d_in[32];
  const float* n4_b    = (const float*)d_in[33];
  const float* mlp_w1  = (const float*)d_in[34];
  const float* mlp_g   = (const float*)d_in[35];
  const float* mlp_b   = (const float*)d_in[36];
  const float* mlp_w2  = (const float*)d_in[37];
  const float* n1_g    = (const float*)d_in[38];
  const float* n1_b    = (const float*)d_in[39];
  float* OUT = (float*)d_out;

  size_t off = 0;
  auto alloc = [&](size_t elems, size_t esz) -> void* {
    void* p = (void*)((char*)d_ws + off);
    off += ((elems * esz + 255) & ~(size_t)255);
    return p;
  };
  u16* PK_PA1 = (u16*)alloc(256 * 64, 2);
  u16* PK_PA2 = (u16*)alloc(64 * 256, 2);
  u16* PK_F1  = (u16*)alloc(64 * 128, 2);
  u16* PK_LA  = (u16*)alloc(9 * 64 * 64, 2);
  u16* PK_F2  = (u16*)alloc(64 * 128, 2);
  u16* PK_P1  = (u16*)alloc(64 * 64, 2);
  u16* PK_C1  = (u16*)alloc(32 * 64, 2);
  u16* PK_C2  = (u16*)alloc(32 * 64, 2);
  u16* PK_CV  = (u16*)alloc(64 * 32, 2);
  u16* PK_P2  = (u16*)alloc(64 * 64, 2);
  u16* PK_M1  = (u16*)alloc(1024 * 256, 2);
  u16* PK_M2  = (u16*)alloc(256 * 1024, 2);

  float* x1_pa  = (float*)alloc(9437184, 4);   // also reused as x4f
  float* xa     = (float*)alloc(37748736, 4);
  float* x2f    = (float*)alloc(9437184, 4);   // also reused as mp3 buffer
  float* tbuf   = (float*)alloc(1048576, 4);
  float* gacc   = (float*)alloc(1048576, 4);
  float* ht     = (float*)alloc(2064384, 4);   // h_transform buf, reused for v
  float* htc    = (float*)alloc(2064384, 4);
  float* x3_mra = (float*)alloc(9437184, 4);
  float* xp     = (float*)alloc(2359296, 4);
  int*   idxb   = (int*)  alloc(2359296, 4);
  float* xg     = (float*)alloc(2359296, 4);
  float* a1     = (float*)alloc(2359296, 4);   // reused as prod
  float* a2     = (float*)alloc(2359296, 4);   // reused as gp2 out
  float* a1c    = (float*)alloc(1179648, 4);
  float* a2c    = (float*)alloc(1179648, 4);
  float* agg    = (float*)alloc(147456, 4);
  float* sig    = (float*)alloc(73728, 4);
  float* amix   = (float*)alloc(1179648, 4);
  float* cvout  = (float*)alloc(2359296, 4);

  dim3 blk(256);
  auto g1 = [&](size_t n) { return dim3((unsigned)((n + 255) / 256)); };
  const long long SB1 = (long long)DSC * PLANE;
  const long long SBF = (long long)CDIM * PLANE;
  const long long SB2 = (long long)DSC * PLANE2;

  // --- weight prepack into WMMA A-fragment layout (bf16) ---
  k_prepack<<<g1(16384), blk, 0, stream>>>(pa_w1, PK_PA1, 256, 64, 1, 0);
  k_prepack<<<g1(16384), blk, 0, stream>>>(pa_w2, PK_PA2, 64, 256, 1, 0);
  k_prepack<<<g1(8192),  blk, 0, stream>>>(fuse1_w, PK_F1, 64, 128, 1, 0);
  for (int tap = 0; tap < 9; ++tap)
    k_prepack<<<g1(4096), blk, 0, stream>>>(la_w, PK_LA + (size_t)tap * 4096, 64, 64, 9, tap);
  k_prepack<<<g1(8192),   blk, 0, stream>>>(fuse2_w, PK_F2, 64, 128, 1, 0);
  k_prepack<<<g1(4096),   blk, 0, stream>>>(g_p1_w, PK_P1, 64, 64, 1, 0);
  k_prepack<<<g1(2048),   blk, 0, stream>>>(g_c1_w, PK_C1, 32, 64, 1, 0);
  k_prepack<<<g1(2048),   blk, 0, stream>>>(g_c2_w, PK_C2, 32, 64, 1, 0);
  k_prepack<<<g1(2048),   blk, 0, stream>>>(g_cv_w, PK_CV, 64, 32, 1, 0);
  k_prepack<<<g1(4096),   blk, 0, stream>>>(g_p2_w, PK_P2, 64, 64, 1, 0);
  k_prepack<<<g1(262144), blk, 0, stream>>>(mlp_w1, PK_M1, 1024, 256, 1, 0);
  k_prepack<<<g1(262144), blk, 0, stream>>>(mlp_w2, PK_M2, 256, 1024, 1, 0);

  // --- branch 1: fused pixel attention (WMMA x2) ---
  k_pa<<<dim3(NPIX / 64), blk, 0, stream>>>(X, PK_PA1, PK_PA2, pa_bn_g, pa_bn_b, x1_pa, xa);

  // --- branch 2: fuse1 (WMMA) then 3x3 conv as 9 WMMA taps + bn + gelu ---
  k_gemm1x1<128><<<dim3(NPIX / 64), blk, 0, stream>>>(x1_pa, SB1, 64, X + (size_t)64 * PLANE,
                                                      SBF, PK_F1, nullptr, 0, x2f, SB1, 64, PLANE);
  k_la<<<dim3(NPIX / 64), blk, 0, stream>>>(x2f, PK_LA, la_bn_g, la_bn_b, xa);

  // --- branch 3: maxpool3 -> blurpool -> strip/diag convs -> gate -> apply ---
  float* mp3buf = x2f;
  k_mp3<<<g1(9437184),  blk, 0, stream>>>(X, mp3buf);
  k_blur<<<g1(1048576), blk, 0, stream>>>(mp3buf, tbuf);
  k_hv1<<<g1(1048576),  blk, 0, stream>>>(tbuf, h1_w, v1_w, gacc);
  k_htr<<<g1(2064384),  blk, 0, stream>>>(tbuf, ht);
  k_hconv<<<g1(2064384), blk, 0, stream>>>(ht, h2_w, htc);
  k_hinv<<<g1(1048576), blk, 0, stream>>>(htc, gacc);
  k_vtr<<<g1(2064384),  blk, 0, stream>>>(tbuf, ht);
  k_vconv<<<g1(2064384), blk, 0, stream>>>(ht, v2_w, htc);
  k_vinv<<<g1(1048576), blk, 0, stream>>>(htc, gacc);
  k_gate<<<g1(1048576), blk, 0, stream>>>(gacc, mra_g, mra_b);
  k_mra<<<g1(9437184),  blk, 0, stream>>>(X, gacc, x3_mra, xa);

  // --- branch 4: fuse2 (WMMA), pooled gated-attention path ---
  float* x4f = x1_pa;
  k_gemm1x1<128><<<dim3(NPIX / 64), blk, 0, stream>>>(x3_mra, SB1, 64, X + (size_t)192 * PLANE,
                                                      SBF, PK_F2, nullptr, 0, x4f, SB1, 64, PLANE);
  k_mp2<<<g1(2359296), blk, 0, stream>>>(x4f, xp, idxb);
  k_gemm1x1<64><<<dim3(NPIX2 / 64), blk, 0, stream>>>(xp, SB2, 64, xp, SB2,
                                                      PK_P1, g_p1_b, 1, xg, SB2, 64, PLANE2);
  k_dw5<<<g1(2359296), blk, 0, stream>>>(xg, g_c0_w, g_c0_b, a1);
  k_dw7<<<g1(2359296), blk, 0, stream>>>(a1, g_cs_w, g_cs_b, a2);
  k_gemm1x1<64><<<dim3(NPIX2 / 64), blk, 0, stream>>>(a1, SB2, 64, a1, SB2,
                                                      PK_C1, g_c1_b, 0, a1c,
                                                      (long long)32 * PLANE2, 32, PLANE2);
  k_gemm1x1<64><<<dim3(NPIX2 / 64), blk, 0, stream>>>(a2, SB2, 64, a2, SB2,
                                                      PK_C2, g_c2_b, 0, a2c,
                                                      (long long)32 * PLANE2, 32, PLANE2);
  k_agg<<<g1(36864),   blk, 0, stream>>>(a1c, a2c, agg);
  k_sq<<<g1(73728),    blk, 0, stream>>>(agg, g_sq_w, g_sq_b, sig);
  k_amix<<<g1(1179648), blk, 0, stream>>>(a1c, a2c, sig, amix);
  k_gemm1x1<32><<<dim3(NPIX2 / 64), blk, 0, stream>>>(amix, (long long)32 * PLANE2, 32, amix,
                                                      (long long)32 * PLANE2,
                                                      PK_CV, g_cv_b, 0, cvout, SB2, 64, PLANE2);
  float* prod = a1;
  k_prod<<<g1(2359296), blk, 0, stream>>>(xg, cvout, prod);
  float* gp2o = a2;
  k_gemm1x1<64><<<dim3(NPIX2 / 64), blk, 0, stream>>>(prod, SB2, 64, prod, SB2,
                                                      PK_P2, g_p2_b, 0, gp2o, SB2, 64, PLANE2);
  k_unpool<<<g1(2359296), blk, 0, stream>>>(gp2o, idxb, X, n4_g, n4_b, xa);

  // --- fused MLP (256->1024->256, hidden in 160KB LDS) + residual -> d_out ---
  k_mlp<<<dim3(NPIX / 64), blk, (64 * 256 + 64 * 1024) * sizeof(u16), stream>>>(
      xa, X, PK_M1, PK_M2, mlp_g, mlp_b, n1_g, n1_b, OUT);
}
